// QuantumInspiredBottleneck_45466523795816
// MI455X (gfx1250) — matrix operand order repaired and checked by reading, not compiled, not run on hardware
//
#include <hip/hip_runtime.h>
#include <hip/hip_bf16.h>
#include <math.h>

typedef __attribute__((ext_vector_type(16))) __bf16 v16bf;
typedef __attribute__((ext_vector_type(8)))  __bf16 bf16x8;
typedef __attribute__((ext_vector_type(8)))  float  f32x8;
typedef __attribute__((ext_vector_type(8)))  float  v8f;

#define B_   128
#define S_   1024
#define DIN  1024
#define NQ_  8
#define NB_  256
#define ROWS (B_ * S_)   // 131072

// ---------------------------------------------------------------------------
// Kernel 3 (runs first): weight pre-conversion to bf16 (stays in L2, 176 KB).
//  Wcb   [16][1024]  : cols 0..7 = amp_w rows, 8..15 = phase_w rows
//  measWb[256][32]   : meas_w^T zero-padded K 8->32, stored K-contiguous per col
//  uncWb [256][256]  : unc_w as-is (row = out col n, contiguous in K)
// ---------------------------------------------------------------------------
__global__ __launch_bounds__(256)
void k_prep(const float* __restrict__ ampW, const float* __restrict__ phW,
            const float* __restrict__ measW, const float* __restrict__ uncW,
            __bf16* __restrict__ Wcb, __bf16* __restrict__ measWb,
            __bf16* __restrict__ uncWb)
{
    int idx = blockIdx.x * 256 + threadIdx.x;
    if (idx < 16384) {
        int c = idx >> 10, k = idx & 1023;
        float v = (c < 8) ? ampW[c * 1024 + k] : phW[(c - 8) * 1024 + k];
        Wcb[c * 1024 + k] = (__bf16)v;
    }
    if (idx < 8192) {
        int n = idx >> 5, k = idx & 31;
        measWb[n * 32 + k] = (__bf16)((k < 8) ? measW[n * 8 + k] : 0.0f);
    }
    if (idx < 65536) uncWb[idx] = (__bf16)uncW[idx];
}

// ---------------------------------------------------------------------------
// Kernel 1: fused amp/phase projection. [131072,1024] x [1024,16] via
// v_wmma_f32_16x16x32_bf16, 32 K-steps per 16-row wave tile. Reads x once
// (512 MB -> memory bound, ~22us floor). Epilogue: bias + tanh / sigmoid*2pi,
// amp<->phase pairing via shfl_xor(8) (C layout: lane = N, vgpr r = M row).
// ---------------------------------------------------------------------------
__global__ __launch_bounds__(256)
void k_qproj(const float* __restrict__ x, const __bf16* __restrict__ Wcb,
             const float* __restrict__ amp_b, const float* __restrict__ phase_b,
             float* __restrict__ R, float* __restrict__ I)
{
    const int lane   = threadIdx.x & 31;
    const int wave   = threadIdx.x >> 5;
    const int lanelo = lane & 15;
    const int half   = lane >> 4;
    const int rowBase = blockIdx.x * 128 + wave * 16;

    const float*  xrow = x   + (size_t)(rowBase + lanelo) * DIN;  // A: row = lane%16
    const __bf16* bcol = Wcb + (size_t)lanelo * DIN;              // B: col = lane%16

    v8f c;
    #pragma unroll
    for (int i = 0; i < 8; ++i) c[i] = 0.0f;

    for (int k0 = 0; k0 < DIN; k0 += 32) {
        // A fragment: elems j -> K = k0 + half*16 + j (contiguous)
        f32x8 x0 = *(const f32x8*)(xrow + k0 + half * 16);
        f32x8 x1 = *(const f32x8*)(xrow + k0 + half * 16 + 8);
        v16bf a;
        #pragma unroll
        for (int j = 0; j < 8; ++j) { a[j] = (__bf16)x0[j]; a[j + 8] = (__bf16)x1[j]; }
        // B fragment: elems j -> K = k0 + half*16 + j (contiguous)
        v16bf b = *(const v16bf*)(bcol + k0 + half * 16);
        c = __builtin_amdgcn_wmma_f32_16x16x32_bf16(false, a, false, b, (short)0, c, false, false);
    }

    const int   col  = lanelo;
    const float bias = (col < 8) ? amp_b[col] : phase_b[col - 8];
    const float TWO_PI = 6.283185307179586f;
    #pragma unroll
    for (int r = 0; r < 8; ++r) {
        float pre   = c[r] + bias;
        float other = __shfl_xor(pre, 8, 32);     // partner column (amp<->phase)
        float amp_pre = (col < 8) ? pre : other;
        float ph_pre  = (col < 8) ? other : pre;
        float a  = tanhf(amp_pre);
        float p  = TWO_PI / (1.0f + __expf(-ph_pre));
        float re = a * __cosf(p);
        float im = a * __sinf(p);
        if (col < 8) {
            int row = rowBase + r + half * 8;
            R[(size_t)row * NQ_ + col] = re;
            I[(size_t)row * NQ_ + col] = im;
        }
    }
}

// ---------------------------------------------------------------------------
// Kernel 2: MHA over L=128 (dim0), one block per sequence position n (1024).
// blockDim = (128, 2): y = real/imag part. Dh=2 -> scalar VALU, two-pass
// softmax (single exp per score). K/V per part staged in LDS. Fuses
// measured = sqrt(re^2 + im^2) after a barrier.
// ---------------------------------------------------------------------------
__global__ __launch_bounds__(256)
void k_mha(const float* __restrict__ R, const float* __restrict__ I,
           const float* __restrict__ inW, const float* __restrict__ inB,
           const float* __restrict__ outW, const float* __restrict__ outB,
           float* __restrict__ M)
{
    __shared__ float Ks[2][128][8];
    __shared__ float Vs[2][128][8];
    __shared__ float resS[2][128][8];
    __shared__ float wIn[192];
    __shared__ float bIn[24];
    __shared__ float wOut[64];
    __shared__ float bOut[8];

    const int n    = blockIdx.x;
    const int l    = threadIdx.x;    // query row 0..127
    const int part = threadIdx.y;    // 0 = real, 1 = imag
    const int t    = part * 128 + l;
    if (t < 192) wIn[t] = inW[t];
    if (t < 24)  bIn[t] = inB[t];
    if (t < 64)  wOut[t] = outW[t];
    if (t < 8)   bOut[t] = outB[t];

    const float* X = part ? I : R;
    float xv[8];
    #pragma unroll
    for (int e = 0; e < 8; ++e) xv[e] = X[((size_t)l * S_ + n) * NQ_ + e];
    __syncthreads();   // weights staged

    float q[8];
    #pragma unroll
    for (int j = 0; j < 8; ++j) {
        float s = bIn[j];
        #pragma unroll
        for (int e = 0; e < 8; ++e) s += wIn[j * 8 + e] * xv[e];
        q[j] = s;
    }
    #pragma unroll
    for (int j = 0; j < 8; ++j) {
        float sk = bIn[8 + j], sv = bIn[16 + j];
        #pragma unroll
        for (int e = 0; e < 8; ++e) {
            sk += wIn[(8 + j) * 8 + e] * xv[e];
            sv += wIn[(16 + j) * 8 + e] * xv[e];
        }
        Ks[part][l][j] = sk; Vs[part][l][j] = sv;
    }
    __syncthreads();

    float o[8];
    const float scale = 0.7071067811865476f;   // 1/sqrt(Dh), Dh=2
    #pragma unroll
    for (int h = 0; h < 4; ++h) {
        float q0 = q[2 * h] * scale, q1 = q[2 * h + 1] * scale;
        float mx = -1e30f;
        for (int m = 0; m < 128; ++m) {
            float s = q0 * Ks[part][m][2 * h] + q1 * Ks[part][m][2 * h + 1];
            mx = fmaxf(mx, s);
        }
        float lsum = 0.0f, a0 = 0.0f, a1 = 0.0f;
        for (int m = 0; m < 128; ++m) {
            float s = q0 * Ks[part][m][2 * h] + q1 * Ks[part][m][2 * h + 1];
            float p = __expf(s - mx);
            lsum += p;
            a0 += p * Vs[part][m][2 * h];
            a1 += p * Vs[part][m][2 * h + 1];
        }
        o[2 * h] = a0 / lsum; o[2 * h + 1] = a1 / lsum;
    }
    #pragma unroll
    for (int e = 0; e < 8; ++e) {
        float s = bOut[e];
        #pragma unroll
        for (int j = 0; j < 8; ++j) s += wOut[e * 8 + j] * o[j];
        resS[part][l][e] = s;
    }
    __syncthreads();
    if (part == 0) {
        #pragma unroll
        for (int e = 0; e < 8; ++e) {
            float er = resS[0][l][e], ei = resS[1][l][e];
            M[((size_t)l * S_ + n) * NQ_ + e] = sqrtf(er * er + ei * ei);
        }
    }
}

// ---------------------------------------------------------------------------
// Kernel 4: bottleneck. Per block: 64 rows x 256 cols, 8 waves.
// Stage 1: compressed = measured[64,8] @ meas_w^T (K zero-padded to 32, one
//          WMMA per N-tile) + bias -> d_out, bf16 copy to LDS.
// Stage 2: uncertainty = sigmoid(compressed @ unc_w^T + bias); K=256.
//          Each 32-wide K-chunk of unc_w (16 KB) is staged into LDS once per
//          block with CDNA5 async copies (global_load_async_to_lds_b128,
//          ASYNCcnt + s_wait_asynccnt), then 8 WMMAs per wave per chunk.
// ---------------------------------------------------------------------------
__global__ __launch_bounds__(256)
void k_bottleneck(const float* __restrict__ M, const __bf16* __restrict__ measWb,
                  const float* __restrict__ measB, const __bf16* __restrict__ uncWb,
                  const float* __restrict__ uncB,
                  float* __restrict__ outC, float* __restrict__ outU)
{
    __shared__ __bf16 compBf[64 * 256];   // 32 KB
    __shared__ __bf16 Bs[256 * 32];       // 16 KB: current K-chunk, [col][32]
    const int t      = threadIdx.x;
    const int lane   = t & 31;
    const int wave   = t >> 5;
    const int lanelo = lane & 15;
    const int half   = lane >> 4;
    const int mtile  = wave >> 1;          // 4 M-tiles of 16 rows
    const int nBase  = (wave & 1) * 8;     // 8 N-tiles per wave
    const int rowBase = blockIdx.x * 64;
    const int lrow   = mtile * 16 + lanelo;
    const int grow   = rowBase + lrow;

    // ---- Stage 1: A = measured (K=8 real, rest zero per ISA A layout) ----
    v16bf a1v;
    #pragma unroll
    for (int j = 0; j < 16; ++j) a1v[j] = (__bf16)0.0f;
    if (half == 0) {   // lo lanes hold K=0..7 in elems 0..7; hi lanes all zero
        f32x8 m = *(const f32x8*)(M + (size_t)grow * NQ_);
        #pragma unroll
        for (int j = 0; j < 8; ++j) a1v[j] = (__bf16)m[j];
    }
    for (int nt = 0; nt < 8; ++nt) {
        const int col = (nBase + nt) * 16 + lanelo;
        v16bf b = *(const v16bf*)(measWb + col * 32 + half * 16);
        v8f c;
        #pragma unroll
        for (int i = 0; i < 8; ++i) c[i] = 0.0f;
        c = __builtin_amdgcn_wmma_f32_16x16x32_bf16(false, a1v, false, b, (short)0, c, false, false);
        const float bias = measB[col];
        #pragma unroll
        for (int r = 0; r < 8; ++r) {
            int lr = mtile * 16 + r + half * 8;
            float v = c[r] + bias;
            outC[(size_t)(rowBase + lr) * NB_ + col] = v;
            compBf[lr * 256 + col] = (__bf16)v;
        }
    }
    __syncthreads();

    // ---- Stage 2: 64x256 GEMM, K=256, async-staged B chunks ----
    v8f acc[8];
    #pragma unroll
    for (int nt = 0; nt < 8; ++nt)
        #pragma unroll
        for (int i = 0; i < 8; ++i) acc[nt][i] = 0.0f;

    for (int k0 = 0; k0 < 256; k0 += 32) {
        // Thread t async-copies unc_w[col=t][k0..k0+32) -> Bs[t][0..32)
        // (64 bytes = 4 x b128; inst offset applies to both LDS and global
        //  addresses per CDNA5 ISA async pseudocode).
        {
            unsigned      ldsoff = (unsigned)(size_t)(&Bs[t * 32]);
            const __bf16* g      = uncWb + (size_t)t * 256 + k0;
            asm volatile(
                "global_load_async_to_lds_b128 %0, %1, off\n\t"
                "global_load_async_to_lds_b128 %0, %1, off offset:16\n\t"
                "global_load_async_to_lds_b128 %0, %1, off offset:32\n\t"
                "global_load_async_to_lds_b128 %0, %1, off offset:48"
                :: "v"(ldsoff), "v"(g) : "memory");
            asm volatile("s_wait_asynccnt 0x0" ::: "memory");
        }
        __syncthreads();   // chunk visible to all waves

        bf16x8 a0 = *(const bf16x8*)&compBf[lrow * 256 + k0 + half * 16];
        bf16x8 a1 = *(const bf16x8*)&compBf[lrow * 256 + k0 + half * 16 + 8];
        v16bf a;
        #pragma unroll
        for (int j = 0; j < 8; ++j) { a[j] = a0[j]; a[j + 8] = a1[j]; }
        #pragma unroll
        for (int nt = 0; nt < 8; ++nt) {
            const int col = (nBase + nt) * 16 + lanelo;
            v16bf b = *(const v16bf*)&Bs[col * 32 + half * 16];
            acc[nt] = __builtin_amdgcn_wmma_f32_16x16x32_bf16(false, a, false, b, (short)0, acc[nt], false, false);
        }
        __syncthreads();   // all reads of Bs done before next chunk overwrite
    }

    #pragma unroll
    for (int nt = 0; nt < 8; ++nt) {
        const int col  = (nBase + nt) * 16 + lanelo;
        const float bias = uncB[col];
        #pragma unroll
        for (int r = 0; r < 8; ++r) {
            int gr = rowBase + mtile * 16 + r + half * 8;
            float v = acc[nt][r] + bias;
            outU[(size_t)gr * NB_ + col] = 1.0f / (1.0f + __expf(-v));
        }
    }
}

// ---------------------------------------------------------------------------
extern "C" void kernel_launch(void* const* d_in, const int* in_sizes, int n_in,
                              void* d_out, int out_size, void* d_ws, size_t ws_size,
                              hipStream_t stream)
{
    (void)in_sizes; (void)n_in; (void)out_size; (void)ws_size;

    const float* x       = (const float*)d_in[0];
    const float* amp_w   = (const float*)d_in[1];
    const float* amp_b   = (const float*)d_in[2];
    const float* phase_w = (const float*)d_in[3];
    const float* phase_b = (const float*)d_in[4];
    const float* in_w    = (const float*)d_in[5];
    const float* in_b    = (const float*)d_in[6];
    const float* out_w   = (const float*)d_in[7];
    const float* out_b   = (const float*)d_in[8];
    const float* meas_w  = (const float*)d_in[9];
    const float* meas_b  = (const float*)d_in[10];
    const float* unc_w   = (const float*)d_in[11];
    const float* unc_b   = (const float*)d_in[12];

    char* ws = (char*)d_ws;
    float*  R      = (float*)(ws);                 // 131072*8 f32 = 4 MB
    float*  I      = (float*)(ws + 4194304);       // 4 MB
    float*  Mbuf   = (float*)(ws + 8388608);       // 4 MB
    __bf16* Wcb    = (__bf16*)(ws + 12582912);     // 32 KB
    __bf16* measWb = (__bf16*)(ws + 12615680);     // 16 KB
    __bf16* uncWb  = (__bf16*)(ws + 12632064);     // 128 KB

    float* outC = (float*)d_out;                   // compressed  [131072,256]
    float* outU = outC + (size_t)ROWS * NB_;       // uncertainty [131072,256]

    k_prep<<<256, 256, 0, stream>>>(amp_w, phase_w, meas_w, unc_w, Wcb, measWb, uncWb);
    k_qproj<<<ROWS / 128, 256, 0, stream>>>(x, Wcb, amp_b, phase_b, R, I);
    k_mha<<<S_, dim3(128, 2, 1), 0, stream>>>(R, I, in_w, in_b, out_w, out_b, Mbuf);
    k_bottleneck<<<ROWS / 64, 256, 0, stream>>>(Mbuf, measWb, meas_b, uncWb, unc_b, outC, outU);
}
